// Attention_65283502899223
// MI455X (gfx1250) — compile-verified, matrix operand order checked by reading, not run on hardware
//
#include <hip/hip_runtime.h>
#include <hip/hip_bf16.h>

#define SEQ   2048
#define HD    128
#define NH    32
#define NKVH  8
#define GQA   4                       // NH / NKVH
#define SCALE 0.08838834764831845f
#define LOG2E 1.4426950408889634f
#define BM    64                      // query rows per workgroup (4 waves x 16)
#define BN    32                      // keys per block
#define KSTR  136                     // lds_k row stride in halves (pad vs 128)
#define VSTR  36                      // lds_vt row stride in halves (8B-aligned rows)

typedef _Float16 v16h __attribute__((ext_vector_type(16)));
typedef _Float16 v8h  __attribute__((ext_vector_type(8)));
typedef _Float16 v4h  __attribute__((ext_vector_type(4)));
typedef float    v8f  __attribute__((ext_vector_type(8)));
typedef float    f4   __attribute__((ext_vector_type(4)));

static __device__ inline v16h combine8(v8h lo, v8h hi) {
    v16h r;
#pragma unroll
    for (int i = 0; i < 8; ++i) { r[i] = lo[i]; r[i + 8] = hi[i]; }
    return r;
}

__global__ __launch_bounds__(128) void fa_gqa_kernel(
    const float* __restrict__ q, const float* __restrict__ k,
    const float* __restrict__ v, float* __restrict__ o)
{
    __shared__ _Float16 lds_k [32 * KSTR];     // K block: [key][d]   f16
    __shared__ _Float16 lds_vt[HD * VSTR];     // V block transposed: [d][key] f16

    const int h      = blockIdx.x;
    const int kvh    = h / GQA;                // repeat_interleave -> h // 4
    const int m_base = blockIdx.y * BM;
    const int tid    = threadIdx.x;
    const int lane   = tid & 31;
    const int wave   = tid >> 5;
    const int hlf    = lane >> 4;
    const int ln     = lane & 15;
    const int w0     = m_base + wave * 16;     // first query row of this wave
    const int qcol   = w0 + ln;                // this lane's query row
    // SGPR copy of the wave-uniform first-row: guarantees scalar branches so
    // EXEC stays all-ones around WMMA (ISA requirement).
    const int w0u    = __builtin_amdgcn_readfirstlane(w0);

    // ---- Q tile into WMMA B-layout f16 registers (lane = query column) ----
    v16h b_q[4];
    {
        const float* qrow = q + (size_t)qcol * (NH * HD) + (size_t)h * HD;
#pragma unroll
        for (int c = 0; c < 4; ++c) {
            const float* base = qrow + c * 32 + hlf * 8;
            f4 f0 = *(const f4*)(base);
            f4 f1 = *(const f4*)(base + 4);
            f4 f2 = *(const f4*)(base + 16);
            f4 f3 = *(const f4*)(base + 20);
            v16h a;
#pragma unroll
            for (int i = 0; i < 4; ++i) {
                a[i]      = (_Float16)f0[i];
                a[4 + i]  = (_Float16)f1[i];
                a[8 + i]  = (_Float16)f2[i];
                a[12 + i] = (_Float16)f3[i];
            }
            b_q[c] = a;
        }
    }

    v8f acc[8];                                // O^T tiles: row=d, col=query
#pragma unroll
    for (int i = 0; i < 8; ++i)
#pragma unroll
        for (int j = 0; j < 8; ++j) acc[i][j] = 0.0f;

    float m_s = -1e30f, l_s = 0.0f;            // per-lane (per-query) stats
    const float SCL = SCALE * LOG2E;           // exp2-domain scale

    const int nkb = (m_base + BM) / BN;        // causal: key blocks needed by WG

    for (int kb = 0; kb < nkb; ++kb) {
        // ---- Cooperative stage: K row-major f16, V transposed f16 ----
#pragma unroll
        for (int t = 0; t < 8; ++t) {
            int linear = tid + t * 128;        // 1024 float4 tiles
            int d4  = linear & 31;             // d = d4*4 (coalesced dim)
            int key = linear >> 5;             // 0..31
            const float* kp = k + (size_t)(kb * 32 + key) * (NKVH * HD) + (size_t)kvh * HD + d4 * 4;
            const float* vp = v + (size_t)(kb * 32 + key) * (NKVH * HD) + (size_t)kvh * HD + d4 * 4;
            f4 kf = *(const f4*)kp;
            f4 vf = *(const f4*)vp;
            v4h kh;
#pragma unroll
            for (int i = 0; i < 4; ++i) kh[i] = (_Float16)kf[i];
            *(v4h*)&lds_k[key * KSTR + d4 * 4] = kh;
#pragma unroll
            for (int i = 0; i < 4; ++i)
                lds_vt[(d4 * 4 + i) * VSTR + key] = (_Float16)vf[i];
        }
        if (kb + 1 < nkb) {                    // global_prefetch next block
            const char* kn = (const char*)(k + (size_t)((kb + 1) * 32 + lane) * (NKVH * HD) + (size_t)kvh * HD);
            const char* vn = (const char*)(v + (size_t)((kb + 1) * 32 + lane) * (NKVH * HD) + (size_t)kvh * HD);
            __builtin_prefetch(kn, 0, 1);
            __builtin_prefetch(vn, 0, 1);
        }
        __syncthreads();

        // Scalar (SGPR) condition: skip blocks fully above diagonal, EXEC untouched
        if (kb * 32 <= w0u + 15) {
            // ---- S^T = K Q^T : software-pipelined A-tile (K) loads ----
            v8f s0, s1;
#pragma unroll
            for (int j = 0; j < 8; ++j) { s0[j] = 0.0f; s1[j] = 0.0f; }
            const _Float16* kb0 = &lds_k[ln * KSTR + hlf * 8];
            const _Float16* kb1 = &lds_k[(16 + ln) * KSTR + hlf * 8];
            v16h ak0 = combine8(*(const v8h*)(kb0), *(const v8h*)(kb0 + 16));
            v16h ak1 = combine8(*(const v8h*)(kb1), *(const v8h*)(kb1 + 16));
#pragma unroll
            for (int c = 0; c < 4; ++c) {
                v16h nk0, nk1;
                if (c < 3) {                   // prefetch next K-dim chunk
                    const _Float16* p0 = kb0 + (c + 1) * 32;
                    const _Float16* p1 = kb1 + (c + 1) * 32;
                    nk0 = combine8(*(const v8h*)(p0), *(const v8h*)(p0 + 16));
                    nk1 = combine8(*(const v8h*)(p1), *(const v8h*)(p1 + 16));
                }
                s0 = __builtin_amdgcn_wmma_f32_16x16x32_f16(false, ak0, false, b_q[c],
                                                            (short)0, s0, false, false);
                s1 = __builtin_amdgcn_wmma_f32_16x16x32_f16(false, ak1, false, b_q[c],
                                                            (short)0, s1, false, false);
                if (c < 3) { ak0 = nk0; ak1 = nk1; }
            }

            // ---- online softmax (exp2 domain); key = kb*32 + {0,16} + 8*hlf + r
            v8f xs0 = s0 * SCL;
            v8f xs1 = s1 * SCL;
            float x0[8], x1[8];
#pragma unroll
            for (int r = 0; r < 8; ++r) { x0[r] = xs0[r]; x1[r] = xs1[r]; }
            if (kb * 32 + 31 > w0u) {          // scalar-branch mask path
                const int key0 = kb * 32 + hlf * 8;
#pragma unroll
                for (int r = 0; r < 8; ++r) {
                    if (key0 + r > qcol)      x0[r] = -1e30f;
                    if (key0 + 16 + r > qcol) x1[r] = -1e30f;
                }
            }
            float mx = fmaxf(x0[0], x1[0]);
#pragma unroll
            for (int r = 1; r < 8; ++r) mx = fmaxf(mx, fmaxf(x0[r], x1[r]));
            mx = fmaxf(mx, __shfl_xor(mx, 16));            // max across the two lane halves
            const float m_new = fmaxf(m_s, mx);

            v16h bp;                                       // P^T already in B-layout
            float rs = 0.0f;
#pragma unroll
            for (int r = 0; r < 8; ++r) {
                float e0 = __builtin_amdgcn_exp2f(x0[r] - m_new);
                float e1 = __builtin_amdgcn_exp2f(x1[r] - m_new);
                rs += e0 + e1;
                bp[r]     = (_Float16)e0;
                bp[8 + r] = (_Float16)e1;
            }
            rs += __shfl_xor(rs, 16);

            if (__ballot(mx > m_s) != 0ull) {              // uniform: rescale needed
                const float corr = __builtin_amdgcn_exp2f(m_s - m_new);
                l_s = l_s * corr + rs;
                m_s = m_new;
#pragma unroll
                for (int t = 0; t < 8; ++t)
                    acc[t] = acc[t] * corr;                // vector mul -> pk/dual
            } else {
                l_s += rs;
            }

            // ---- O^T += V^T P^T : 2-deep software-pipelined A-tile loads ----
            const _Float16* vb = &lds_vt[ln * VSTR + hlf * 8];
#define LDV(T, DST)                                                     \
            {                                                           \
                const _Float16* vp_ = vb + (T) * 16 * VSTR;             \
                v4h a0 = *(const v4h*)(vp_);                            \
                v4h a1 = *(const v4h*)(vp_ + 4);                        \
                v4h b0 = *(const v4h*)(vp_ + 16);                       \
                v4h b1 = *(const v4h*)(vp_ + 20);                       \
                _Pragma("unroll")                                       \
                for (int i = 0; i < 4; ++i) {                           \
                    DST[i] = a0[i]; DST[4 + i] = a1[i];                 \
                    DST[8 + i] = b0[i]; DST[12 + i] = b1[i];            \
                }                                                       \
            }
            v16h av0, av1;
            LDV(0, av0);
            LDV(1, av1);
#pragma unroll
            for (int t = 0; t < 8; ++t) {
                v16h nx;
                if (t < 6) LDV(t + 2, nx);     // keep 2 tiles in flight
                acc[t] = __builtin_amdgcn_wmma_f32_16x16x32_f16(false, av0, false, bp,
                                                                (short)0, acc[t], false, false);
                av0 = av1;
                if (t < 6) av1 = nx;
            }
#undef LDV
        }
        __syncthreads();
    }

    // ---- Epilogue: one query row per lane, 8 consecutive d per tile ----
    const float inv = 1.0f / l_s;
    float* op = o + (size_t)qcol * (NH * HD) + (size_t)h * HD + hlf * 8;
#pragma unroll
    for (int t = 0; t < 8; ++t) {
        v8f sc = acc[t] * inv;
        f4 lo = __builtin_shufflevector(sc, sc, 0, 1, 2, 3);
        f4 hi = __builtin_shufflevector(sc, sc, 4, 5, 6, 7);
        *(f4*)(op + t * 16)     = lo;
        *(f4*)(op + t * 16 + 4) = hi;
    }
}

extern "C" void kernel_launch(void* const* d_in, const int* in_sizes, int n_in,
                              void* d_out, int out_size, void* d_ws, size_t ws_size,
                              hipStream_t stream) {
    const float* q = (const float*)d_in[0];
    const float* k = (const float*)d_in[1];
    const float* v = (const float*)d_in[2];
    float* o = (float*)d_out;
    dim3 grid(NH, SEQ / BM);   // (head, query-block)
    dim3 block(128);           // 4 waves of 32
    fa_gqa_kernel<<<grid, block, 0, stream>>>(q, k, v, o);
}
